// RotaryMultiHeadAttention_27101243638365
// MI455X (gfx1250) — matrix operand order repaired and checked by reading, not compiled, not run on hardware
//
#include <hip/hip_runtime.h>
#include <hip/hip_bf16.h>
#include <math.h>

// ---------------------------------------------------------------------------
// RotaryMultiHeadAttention for MI455X (gfx1250), wave32 + WMMA bf16 pipeline.
// TDM (tensor_load_to_lds) stages the shared operand of every GEMM stage
// (B panels in projections, K/V tiles in flash attention) into double-
// buffered LDS, synchronized with TENSORcnt + workgroup barriers.
// B=2, N=2048, d_model=1024, H=16, Dh=64.
// ---------------------------------------------------------------------------

#define BATCH    2
#define SEQ      2048
#define DMODEL   1024
#define HEADS    16
#define HDIM     64
#define MTOT     (BATCH * SEQ)          // 4096 tokens
#define ROPE_SCALE_F (6.2831853071795864769f / 1024.0f)

typedef __attribute__((ext_vector_type(16))) __bf16 v16bf;
typedef __attribute__((ext_vector_type(8)))  float  v8f;
typedef __attribute__((ext_vector_type(4)))  unsigned int v4u;
typedef __attribute__((ext_vector_type(8)))  int    v8i;
typedef __attribute__((ext_vector_type(4)))  int    v4i;

#if defined(__has_builtin)
#  if __has_builtin(__builtin_amdgcn_tensor_load_to_lds)
#    define USE_TDM 1
#  endif
#endif
#ifndef USE_TDM
#  define USE_TDM 0
#endif

#if defined(__AMDGCN__)
#  if USE_TDM
#    warning "CDNA5 device pass: TDM tensor_load_to_lds path ENABLED"
#  else
#    warning "CDNA5 device pass: TDM path DISABLED (cooperative-load fallback)"
#  endif
#endif

union AFrag {
    v16bf v;
    uint4 q[2];
    unsigned short s[16];
};

__device__ __forceinline__ unsigned short f32_to_bf16(float f) {
    unsigned u = __float_as_uint(f);
    unsigned r = u + 0x7FFFu + ((u >> 16) & 1u);   // round-to-nearest-even
    return (unsigned short)(r >> 16);
}

__device__ __forceinline__ v8f wmma_bf16(const AFrag& a, const AFrag& b, v8f c) {
    // (neg_a, A, neg_b, B, c_mod, C, reuse_a, reuse_b)
    return __builtin_amdgcn_wmma_f32_16x16x32_bf16(false, a.v, false, b.v,
                                                   (short)0, c, false, false);
}

#if USE_TDM
// Issue a 2-D TDM load: tile (tile1 lines x tile0 elems) of bf16 from a
// row-major tensor (line length tdim0, line stride stride0, tdim1 lines)
// into LDS at byte offset lds_off. D# layout per CDNA5 ISA 8.3/8.4.
__device__ __forceinline__ void tdm_load_2d(unsigned int lds_off,
                                            const void* gaddr,
                                            unsigned int tdim0,
                                            unsigned int tdim1,
                                            unsigned int tile0,
                                            unsigned int tile1,
                                            unsigned int stride0)
{
    const unsigned long long ga = (unsigned long long)(uintptr_t)gaddr;
    v4u g0;
    g0[0] = 1u;                                   // count=1, user descriptor
    g0[1] = lds_off;                              // lds_addr
    g0[2] = (unsigned int)ga;                     // global_addr[31:0]
    g0[3] = (unsigned int)(ga >> 32) | (2u << 30);// global_addr[56:32] | type=2
    v8i g1;
    g1[0] = (int)(1u << 16);                      // wg_mask=0, data_size=1 (2B)
    g1[1] = (int)((tdim0 & 0xFFFFu) << 16);       // tensor_dim0[15:0]
    g1[2] = (int)((tdim0 >> 16) | ((tdim1 & 0xFFFFu) << 16));
    g1[3] = (int)((tdim1 >> 16) | (tile0 << 16)); // tile_dim0
    g1[4] = (int)(tile1 & 0xFFFFu);               // tile_dim1 (tile_dim2 = 0)
    g1[5] = (int)stride0;                         // tensor_dim0_stride[31:0]
    g1[6] = 0;
    g1[7] = 0;
    v4i z4 = {0, 0, 0, 0};
#if defined(__clang_major__) && (__clang_major__ >= 23)
    v8i z8 = {0, 0, 0, 0, 0, 0, 0, 0};
    __builtin_amdgcn_tensor_load_to_lds(g0, g1, z4, z4, z8, 0);
#else
    __builtin_amdgcn_tensor_load_to_lds(g0, g1, z4, z4, 0);
#endif
}
#endif

// ---------------------------------------------------------------------------
// fp32 -> bf16 bulk convert
// ---------------------------------------------------------------------------
__global__ void cvt_f32_bf16(const float* __restrict__ src,
                             unsigned short* __restrict__ dst, int n) {
    int i = blockIdx.x * blockDim.x + threadIdx.x;
    if (i < n) dst[i] = f32_to_bf16(src[i]);
}

// ---------------------------------------------------------------------------
// Fused QKV projection: out[tok, o] = sum_d xb[tok,d] * W[o,d] + bias[o]
// Block = 8 waves sharing one 64-column B panel (256 rows x 64 cols total);
// each wave owns a 32x64 tile. Per 64-deep K chunk, the TDM stages a
// 64x64 bf16 B tile into double-buffered LDS (wave 0 issues, TENSORcnt
// synced), overlapping DMA of chunk c+1 with 16 WMMAs on chunk c.
// Q,K written fp32 [B,H,N,64] (RoPE next); V written bf16 transposed
// [B,H,64,N] so attention P*V B-fragments are contiguous.
// ---------------------------------------------------------------------------
__global__ __launch_bounds__(256) void qkv_gemm(
    const unsigned short* __restrict__ xb,    // [4096,1024] bf16
    const unsigned short* __restrict__ Wqb,   // [1024,1024] bf16
    const unsigned short* __restrict__ Wkb,
    const unsigned short* __restrict__ Wvb,
    const float* __restrict__ bq, const float* __restrict__ bk,
    const float* __restrict__ bv,
    float* __restrict__ Qf, float* __restrict__ Kf,   // [B,H,N,64] fp32
    unsigned short* __restrict__ Vt)                  // [B,H,64,N] bf16
{
    __shared__ __align__(16) unsigned short Bt[2][64][64];   // [col][k] 16KB

    const int lane = threadIdx.x & 31;
    const int wave = threadIdx.x >> 5;
    const int mblk = blockIdx.x / 48;             // 16 blocks of 256 rows
    const int ct   = blockIdx.x % 48;
    const int ocat = (ct * 64) >> 10;             // 0=Q 1=K 2=V
    const int col0 = (ct * 64) & 1023;            // column within W
    const unsigned short* W = (ocat == 0) ? Wqb : (ocat == 1) ? Wkb : Wvb;
    const float* bias       = (ocat == 0) ? bq  : (ocat == 1) ? bk  : bv;

    const int hi   = lane >> 4;                   // half-wave: 0/1
    const int l16  = lane & 15;
    const int aklo = hi * 8;                      // A K-offset per layout
    const int bklo = hi * 8;                      // B K-offset per layout

    const int row0 = mblk * 256 + wave * 32;      // wave's 32-row tile
    const unsigned short* arp0 = xb + (size_t)(row0 + l16) * DMODEL;
    const unsigned short* arp1 = arp0 + (size_t)16 * DMODEL;

    auto stage_issue = [&](int buf, int k0) {
#if USE_TDM
        if (wave == 0) {
            tdm_load_2d((unsigned int)(uintptr_t)&Bt[buf][0][0],
                        W + (size_t)col0 * DMODEL + k0,
                        /*tdim0=*/DMODEL, /*tdim1=*/DMODEL,
                        /*tile0=*/64, /*tile1=*/64, /*stride0=*/DMODEL);
        }
#else
        const int t = threadIdx.x;
        const int r = t >> 2;             // B row (output col) 0..63
        const int c = (t & 3) * 16;       // k offset 0,16,32,48
        *(uint4*)&Bt[buf][r][c] =
            *(const uint4*)(W + (size_t)(col0 + r) * DMODEL + k0 + c);
        *(uint4*)&Bt[buf][r][c + 8] =
            *(const uint4*)(W + (size_t)(col0 + r) * DMODEL + k0 + c + 8);
#endif
    };
    auto stage_wait = [&]() {
#if USE_TDM
        __syncthreads();
        if (wave == 0) __builtin_amdgcn_s_wait_tensorcnt(0);
        __syncthreads();
#else
        __syncthreads();
#endif
    };

    v8f acc[2][4] = {};
    stage_issue(0, 0);
    stage_wait();

    for (int cc = 0; cc < DMODEL / 64; ++cc) {
        const int cur = cc & 1;
        if (cc + 1 < DMODEL / 64) stage_issue(cur ^ 1, (cc + 1) * 64);
        const int kbase = cc * 64;
#pragma unroll
        for (int ks = 0; ks < 2; ++ks) {
            const int k0 = kbase + ks * 32;
            AFrag a0, a1;
            a0.q[0] = *(const uint4*)(arp0 + k0 + aklo);
            a0.q[1] = *(const uint4*)(arp0 + k0 + 16 + aklo);
            a1.q[0] = *(const uint4*)(arp1 + k0 + aklo);
            a1.q[1] = *(const uint4*)(arp1 + k0 + 16 + aklo);
            __builtin_prefetch(arp0 + k0 + 256, 0, 1);
            __builtin_prefetch(arp1 + k0 + 256, 0, 1);
#pragma unroll
            for (int sub = 0; sub < 4; ++sub) {
                const unsigned short* bp =
                    &Bt[cur][sub * 16 + l16][ks * 32 + bklo];
                AFrag b;
                b.q[0] = *(const uint4*)(bp);
                b.q[1] = *(const uint4*)(bp + 16);
                acc[0][sub] = wmma_bf16(a0, b, acc[0][sub]);
                acc[1][sub] = wmma_bf16(a1, b, acc[1][sub]);
            }
        }
        stage_wait();
    }

    float bcol[4];
#pragma unroll
    for (int sub = 0; sub < 4; ++sub) bcol[sub] = bias[col0 + sub * 16 + l16];

#pragma unroll
    for (int ms = 0; ms < 2; ++ms) {
#pragma unroll
        for (int i = 0; i < 8; ++i) {
            const int tok = row0 + ms * 16 + i + hi * 8;
            const int bb  = tok >> 11;                // /SEQ
            const int nn  = tok & (SEQ - 1);
#pragma unroll
            for (int sub = 0; sub < 4; ++sub) {
                const int o  = col0 + sub * 16 + l16; // D layout: N = lane&15
                const float val = acc[ms][sub][i] + bcol[sub];
                const int h  = o >> 6;
                const int dd = o & 63;
                if (ocat == 2) {
                    Vt[(((size_t)bb * HEADS + h) * HDIM + dd) * SEQ + nn] =
                        f32_to_bf16(val);
                } else {
                    float* dst = (ocat == 0) ? Qf : Kf;
                    dst[(((size_t)bb * HEADS + h) * SEQ + nn) * HDIM + dd] = val;
                }
            }
        }
    }
}

// ---------------------------------------------------------------------------
// 2D RoPE on fp32 [B,H,N,64] -> bf16 [B,H,N,64]. One thread per rotated pair.
// ---------------------------------------------------------------------------
__global__ void rope_cvt(const float* __restrict__ src,
                         const float* __restrict__ pos,    // [B,N,2]
                         unsigned short* __restrict__ dst)
{
    const size_t idx = (size_t)blockIdx.x * blockDim.x + threadIdx.x;
    const size_t total = (size_t)BATCH * HEADS * SEQ * 32;
    if (idx >= total) return;
    const int p    = idx & 31;          // pair id 0..31
    const size_t t = idx >> 5;          // (b*H + h)*SEQ + n
    const int n    = (int)(t & (SEQ - 1));
    const int b    = (int)(t >> 11) >> 4;
    const int plane = p >> 4;           // 0 = x-plane, 1 = y-plane
    const int pi    = p & 15;           // pair within plane
    const float posv = pos[((size_t)b * SEQ + n) * 2 + plane];
    const float inv  = powf(10000.0f, -(2.0f * (float)pi) / 32.0f);
    const float ang  = posv * ROPE_SCALE_F * inv;
    const float s = sinf(ang), c = cosf(ang);
    const size_t base = t * HDIM + plane * 32 + 2 * pi;
    const float x0 = src[base], x1 = src[base + 1];
    dst[base]     = f32_to_bf16(x0 * c - x1 * s);
    dst[base + 1] = f32_to_bf16(x0 * s + x1 * c);
}

// ---------------------------------------------------------------------------
// Flash attention: 8 waves/block share one (b,h); each wave owns a 16-row
// q tile (block covers 128 q rows). K/V 32-key tiles are staged into
// double-buffered LDS by the Tensor Data Mover (wave 0 issues, TENSORcnt
// synchronized), overlapping DMA of tile j+1 with WMMA compute on tile j.
// S = Q.K^T via 4 wmma, online softmax (16-lane shfl reductions),
// P transposed C->A layout through LDS, O += P.V via 4 wmma.
// ---------------------------------------------------------------------------
__global__ __launch_bounds__(256) void flash_attn(
    const unsigned short* __restrict__ Qb,   // [B,H,N,64] bf16 (post-RoPE)
    const unsigned short* __restrict__ Kb,   // [B,H,N,64] bf16 (post-RoPE)
    const unsigned short* __restrict__ Vt,   // [B,H,64,N] bf16
    unsigned short* __restrict__ Ob)         // [4096,1024] bf16
{
    __shared__ __align__(16) unsigned short Kt[2][32][64];    // [key][d]
    __shared__ __align__(16) unsigned short Vtl[2][64][32];   // [d][key]
    __shared__ __align__(16) unsigned short pbuf[8][16][32];

    const int lane = threadIdx.x & 31;
    const int wave = threadIdx.x >> 5;
    const int bh   = blockIdx.x >> 4;         // (b*16 + h), 32 heads
    const int qblk = blockIdx.x & 15;         // 16 q-blocks of 128 rows
    const int qt   = qblk * 8 + wave;         // this wave's 16-row q tile
    const int h    = bh & 15;
    const int b    = bh >> 4;
    const int hi   = lane >> 4;
    const int l16  = lane & 15;

    const unsigned short* Qhead = Qb + (size_t)bh * SEQ * HDIM;
    const unsigned short* Khead = Kb + (size_t)bh * SEQ * HDIM;
    const unsigned short* Vhead = Vt + (size_t)bh * HDIM * SEQ;

    // Q A-fragments: d 0..31 and 32..63
    AFrag qf0, qf1;
    {
        const unsigned short* qp = Qhead + (size_t)(qt * 16 + l16) * HDIM;
        const int aklo = hi * 16;
        qf0.q[0] = *(const uint4*)(qp + 0  + aklo);
        qf0.q[1] = *(const uint4*)(qp + 8  + aklo);
        qf1.q[0] = *(const uint4*)(qp + 32 + aklo);
        qf1.q[1] = *(const uint4*)(qp + 40 + aklo);
    }

    float mrow[8], lrow[8];
#pragma unroll
    for (int i = 0; i < 8; ++i) { mrow[i] = -3.0e38f; lrow[i] = 0.0f; }
    v8f oacc[4] = {};
    const float scale = 0.125f;   // 1/sqrt(64)

    // ---- K/V tile staging ------------------------------------------------
    auto stage_issue = [&](int buf, int jb) {
#if USE_TDM
        if (wave == 0) {
            tdm_load_2d((unsigned int)(uintptr_t)&Kt[buf][0][0],
                        Khead + (size_t)jb * HDIM,
                        /*tdim0=*/HDIM, /*tdim1=*/SEQ,
                        /*tile0=*/HDIM, /*tile1=*/32, /*stride0=*/HDIM);
            tdm_load_2d((unsigned int)(uintptr_t)&Vtl[buf][0][0],
                        Vhead + jb,
                        /*tdim0=*/SEQ, /*tdim1=*/HDIM,
                        /*tile0=*/32, /*tile1=*/HDIM, /*stride0=*/SEQ);
        }
#else
        // Cooperative fallback: 256 threads x 8 bf16 each per tile.
        {
            const int t  = threadIdx.x;
            const int kr = t >> 3;              // key row 0..31
            const int kc = (t & 7) * 8;         // d col
            *(uint4*)&Kt[buf][kr][kc] =
                *(const uint4*)(Khead + (size_t)(jb + kr) * HDIM + kc);
            const int vr = t >> 2;              // d row 0..63
            const int vc = (t & 3) * 8;         // key col
            *(uint4*)&Vtl[buf][vr][vc] =
                *(const uint4*)(Vhead + (size_t)vr * SEQ + jb + vc);
        }
#endif
    };
    auto stage_wait = [&]() {
#if USE_TDM
        __syncthreads();   // all waves done reading the buffer being replaced
        if (wave == 0) __builtin_amdgcn_s_wait_tensorcnt(0);
        __syncthreads();   // staged data visible to whole workgroup
#else
        __syncthreads();
#endif
    };

    stage_issue(0, 0);
    stage_wait();

    for (int j = 0; j < SEQ / 32; ++j) {
        const int cur = j & 1;
        if (j + 1 < SEQ / 32) stage_issue(cur ^ 1, (j + 1) * 32);

        // ---- S = Q.K^T over this 32-key tile (from LDS) ----
        v8f s0 = {}, s1 = {};
        {
            const unsigned short* kp = &Kt[cur][l16][hi * 16];
            AFrag b0, b1;
            b0.q[0] = *(const uint4*)(kp);
            b0.q[1] = *(const uint4*)(kp + 8);
            b1.q[0] = *(const uint4*)(kp + 32);
            b1.q[1] = *(const uint4*)(kp + 40);
            s0 = wmma_bf16(qf0, b0, s0);
            s0 = wmma_bf16(qf1, b1, s0);
        }
        {
            const unsigned short* kp = &Kt[cur][16 + l16][hi * 16];
            AFrag b0, b1;
            b0.q[0] = *(const uint4*)(kp);
            b0.q[1] = *(const uint4*)(kp + 8);
            b1.q[0] = *(const uint4*)(kp + 32);
            b1.q[1] = *(const uint4*)(kp + 40);
            s1 = wmma_bf16(qf0, b0, s1);
            s1 = wmma_bf16(qf1, b1, s1);
        }

        // ---- online softmax per row (row m = i + hi*8; cols across lanes)
#pragma unroll
        for (int i = 0; i < 8; ++i) {
            float a0 = s0[i] * scale, a1 = s1[i] * scale;
            float t = fmaxf(a0, a1);
            t = fmaxf(t, __shfl_xor(t, 1, 16));
            t = fmaxf(t, __shfl_xor(t, 2, 16));
            t = fmaxf(t, __shfl_xor(t, 4, 16));
            t = fmaxf(t, __shfl_xor(t, 8, 16));
            const float mn    = fmaxf(mrow[i], t);
            const float alpha = __expf(mrow[i] - mn);
            const float p0 = __expf(a0 - mn);
            const float p1 = __expf(a1 - mn);
            float sum = p0 + p1;
            sum += __shfl_xor(sum, 1, 16);
            sum += __shfl_xor(sum, 2, 16);
            sum += __shfl_xor(sum, 4, 16);
            sum += __shfl_xor(sum, 8, 16);
            lrow[i] = lrow[i] * alpha + sum;
            mrow[i] = mn;
#pragma unroll
            for (int c2 = 0; c2 < 4; ++c2) oacc[c2][i] *= alpha;
            pbuf[wave][i + hi * 8][l16]      = f32_to_bf16(p0);
            pbuf[wave][i + hi * 8][16 + l16] = f32_to_bf16(p1);
        }
        asm volatile("s_wait_dscnt 0" ::: "memory");   // DS in-order per wave

        // ---- P back out in A-layout (16 rows x 32 keys) ----
        AFrag pf;
        {
            const unsigned short* pp = &pbuf[wave][l16][0];
            const int klo = hi * 16;
            pf.q[0] = *(const uint4*)(pp + klo);
            pf.q[1] = *(const uint4*)(pp + klo + 8);
        }
        // ---- O += P.V (4 chunks of 16 output dims, V from LDS) ----
#pragma unroll
        for (int c2 = 0; c2 < 4; ++c2) {
            const unsigned short* vp = &Vtl[cur][c2 * 16 + l16][hi * 16];
            AFrag vf;
            vf.q[0] = *(const uint4*)(vp);
            vf.q[1] = *(const uint4*)(vp + 8);
            oacc[c2] = wmma_bf16(pf, vf, oacc[c2]);
        }

        stage_wait();
    }

    // normalize and emit bf16 attention output
#pragma unroll
    for (int i = 0; i < 8; ++i) {
        const size_t tok = (size_t)b * SEQ + qt * 16 + i + hi * 8;
        const float inv_l = 1.0f / lrow[i];
#pragma unroll
        for (int c2 = 0; c2 < 4; ++c2) {
            const int o = h * HDIM + c2 * 16 + l16;
            Ob[tok * DMODEL + o] = f32_to_bf16(oacc[c2][i] * inv_l);
        }
    }
}

// ---------------------------------------------------------------------------
// Output projection: d_out[tok,o] = sum_d Ob[tok,d]*Wo[o,d] + bo[o]  (fp32)
// Same TDM-staged shared-B structure as qkv_gemm (block = 256 rows x 64 cols).
// ---------------------------------------------------------------------------
__global__ __launch_bounds__(256) void out_proj(
    const unsigned short* __restrict__ Ob,    // [4096,1024] bf16
    const unsigned short* __restrict__ Wob,   // [1024,1024] bf16
    const float* __restrict__ bo,
    float* __restrict__ out)                  // [4096,1024] fp32
{
    __shared__ __align__(16) unsigned short Bt[2][64][64];   // [col][k] 16KB

    const int lane = threadIdx.x & 31;
    const int wave = threadIdx.x >> 5;
    const int mblk = blockIdx.x >> 4;             // 16 blocks of 256 rows
    const int col0 = (blockIdx.x & 15) * 64;

    const int hi   = lane >> 4;
    const int l16  = lane & 15;
    const int aklo = hi * 8;
    const int bklo = hi * 8;

    const int row0 = mblk * 256 + wave * 32;
    const unsigned short* arp0 = Ob + (size_t)(row0 + l16) * DMODEL;
    const unsigned short* arp1 = arp0 + (size_t)16 * DMODEL;

    auto stage_issue = [&](int buf, int k0) {
#if USE_TDM
        if (wave == 0) {
            tdm_load_2d((unsigned int)(uintptr_t)&Bt[buf][0][0],
                        Wob + (size_t)col0 * DMODEL + k0,
                        /*tdim0=*/DMODEL, /*tdim1=*/DMODEL,
                        /*tile0=*/64, /*tile1=*/64, /*stride0=*/DMODEL);
        }
#else
        const int t = threadIdx.x;
        const int r = t >> 2;
        const int c = (t & 3) * 16;
        *(uint4*)&Bt[buf][r][c] =
            *(const uint4*)(Wob + (size_t)(col0 + r) * DMODEL + k0 + c);
        *(uint4*)&Bt[buf][r][c + 8] =
            *(const uint4*)(Wob + (size_t)(col0 + r) * DMODEL + k0 + c + 8);
#endif
    };
    auto stage_wait = [&]() {
#if USE_TDM
        __syncthreads();
        if (wave == 0) __builtin_amdgcn_s_wait_tensorcnt(0);
        __syncthreads();
#else
        __syncthreads();
#endif
    };

    v8f acc[2][4] = {};
    stage_issue(0, 0);
    stage_wait();

    for (int cc = 0; cc < DMODEL / 64; ++cc) {
        const int cur = cc & 1;
        if (cc + 1 < DMODEL / 64) stage_issue(cur ^ 1, (cc + 1) * 64);
        const int kbase = cc * 64;
#pragma unroll
        for (int ks = 0; ks < 2; ++ks) {
            const int k0 = kbase + ks * 32;
            AFrag a0, a1;
            a0.q[0] = *(const uint4*)(arp0 + k0 + aklo);
            a0.q[1] = *(const uint4*)(arp0 + k0 + 16 + aklo);
            a1.q[0] = *(const uint4*)(arp1 + k0 + aklo);
            a1.q[1] = *(const uint4*)(arp1 + k0 + 16 + aklo);
            __builtin_prefetch(arp0 + k0 + 256, 0, 1);
            __builtin_prefetch(arp1 + k0 + 256, 0, 1);
#pragma unroll
            for (int sub = 0; sub < 4; ++sub) {
                const unsigned short* bp =
                    &Bt[cur][sub * 16 + l16][ks * 32 + bklo];
                AFrag b;
                b.q[0] = *(const uint4*)(bp);
                b.q[1] = *(const uint4*)(bp + 16);
                acc[0][sub] = wmma_bf16(a0, b, acc[0][sub]);
                acc[1][sub] = wmma_bf16(a1, b, acc[1][sub]);
            }
        }
        stage_wait();
    }

    float bcol[4];
#pragma unroll
    for (int sub = 0; sub < 4; ++sub) bcol[sub] = bo[col0 + sub * 16 + l16];

#pragma unroll
    for (int ms = 0; ms < 2; ++ms) {
#pragma unroll
        for (int i = 0; i < 8; ++i) {
            const size_t tok = (size_t)row0 + ms * 16 + i + hi * 8;
#pragma unroll
            for (int sub = 0; sub < 4; ++sub) {
                const int o = col0 + sub * 16 + l16;
                out[tok * DMODEL + o] = acc[ms][sub][i] + bcol[sub];
            }
        }
    }
}

// ---------------------------------------------------------------------------
// Host-side orchestration
// ---------------------------------------------------------------------------
extern "C" void kernel_launch(void* const* d_in, const int* in_sizes, int n_in,
                              void* d_out, int out_size, void* d_ws, size_t ws_size,
                              hipStream_t stream) {
    const float* x    = (const float*)d_in[0];
    const float* qpos = (const float*)d_in[1];
    const float* kpos = (const float*)d_in[2];
    const float* Wq   = (const float*)d_in[3];
    const float* bq   = (const float*)d_in[4];
    const float* Wk   = (const float*)d_in[5];
    const float* bk   = (const float*)d_in[6];
    const float* Wv   = (const float*)d_in[7];
    const float* bv   = (const float*)d_in[8];
    const float* Wo   = (const float*)d_in[9];
    const float* bo   = (const float*)d_in[10];

    char* ws = (char*)d_ws;
    size_t off = 0;
    auto alloc = [&](size_t bytes) -> void* {
        void* p = ws + off;
        off = (off + bytes + 255) & ~(size_t)255;
        return p;
    };

    unsigned short* xb  = (unsigned short*)alloc((size_t)MTOT * DMODEL * 2);
    unsigned short* Wqb = (unsigned short*)alloc((size_t)DMODEL * DMODEL * 2);
    unsigned short* Wkb = (unsigned short*)alloc((size_t)DMODEL * DMODEL * 2);
    unsigned short* Wvb = (unsigned short*)alloc((size_t)DMODEL * DMODEL * 2);
    unsigned short* Wob = (unsigned short*)alloc((size_t)DMODEL * DMODEL * 2);
    float*          Qf  = (float*)alloc((size_t)MTOT * DMODEL * 4);
    float*          Kf  = (float*)alloc((size_t)MTOT * DMODEL * 4);
    unsigned short* Qb  = (unsigned short*)alloc((size_t)MTOT * DMODEL * 2);
    unsigned short* Kb  = (unsigned short*)alloc((size_t)MTOT * DMODEL * 2);
    unsigned short* Vt  = (unsigned short*)alloc((size_t)MTOT * DMODEL * 2);
    unsigned short* Ob  = (unsigned short*)alloc((size_t)MTOT * DMODEL * 2);

    // 1) convert inputs to bf16
    const int nx = MTOT * DMODEL;        // 4,194,304
    const int nw = DMODEL * DMODEL;      // 1,048,576
    cvt_f32_bf16<<<(nx + 255) / 256, 256, 0, stream>>>(x,  xb,  nx);
    cvt_f32_bf16<<<(nw + 255) / 256, 256, 0, stream>>>(Wq, Wqb, nw);
    cvt_f32_bf16<<<(nw + 255) / 256, 256, 0, stream>>>(Wk, Wkb, nw);
    cvt_f32_bf16<<<(nw + 255) / 256, 256, 0, stream>>>(Wv, Wvb, nw);
    cvt_f32_bf16<<<(nw + 255) / 256, 256, 0, stream>>>(Wo, Wob, nw);

    // 2) fused QKV projection (WMMA + TDM-staged B): 16 mblocks * 48 ctiles
    qkv_gemm<<<768, 256, 0, stream>>>(xb, Wqb, Wkb, Wvb,
                                      bq, bk, bv, Qf, Kf, Vt);

    // 3) RoPE + bf16 convert for Q and K
    const int nrope = BATCH * HEADS * SEQ * 32;   // 2,097,152
    rope_cvt<<<(nrope + 255) / 256, 256, 0, stream>>>(Qf, qpos, Qb);
    rope_cvt<<<(nrope + 255) / 256, 256, 0, stream>>>(Kf, kpos, Kb);

    // 4) flash attention (WMMA + TDM-staged K/V): 512 blocks x 8 waves
    flash_attn<<<512, 256, 0, stream>>>(Qb, Kb, Vt, Ob);

    // 5) output projection (WMMA + TDM-staged B): 16 mblocks * 16 ctiles
    out_proj<<<256, 256, 0, stream>>>(Ob, Wob, bo, (float*)d_out);

    (void)in_sizes; (void)n_in; (void)out_size; (void)ws_size;
}
